// ADASYNNET_83288005804284
// MI455X (gfx1250) — matrix operand order repaired and checked by reading, not compiled, not run on hardware
//
#include <hip/hip_runtime.h>
#include <stdint.h>

// CDNA5 WMMA vector types
typedef __attribute__((ext_vector_type(16))) _Float16 v16h;
typedef __attribute__((ext_vector_type(8)))  float    v8f;

#define T_STEPS   200
#define T_HALF    100
#define BATCH     128
#define NFEAT     10000
#define HOUT      3
#define HALF_COUNT 128000000u      // T_HALF * BATCH * NFEAT  (jax iota split point)

#define N_SEG           8          // K-dimension segments (occupancy)
#define CHUNKS          313        // ceil(10000 / 32)
#define FULL_CHUNKS     312        // chunks with all 32 K-values in bounds
#define CHUNKS_PER_SEG  40         // ceil(313 / 8)
#define PAIRS           12800      // T_HALF * BATCH
#define WAVES_PER_BLOCK 8

#define F16_ONE_LO 0x00003c00u     // f16 1.0 in low half
#define F16_ONE_HI 0x3c000000u     // f16 1.0 in high half

__device__ __forceinline__ uint32_t rotl32(uint32_t x, uint32_t d) {
  return (x << d) | (x >> (32u - d));   // lowers to v_alignbit_b32
}

// JAX threefry2x32, 20 rounds, key = (0, 42)  [jax.random.key(42)]
__device__ __forceinline__ void threefry2x32(uint32_t c0, uint32_t c1,
                                             uint32_t &o0, uint32_t &o1) {
  const uint32_t ks0 = 0u;
  const uint32_t ks1 = 42u;
  const uint32_t ks2 = 0x1BD11BDAu ^ 0u ^ 42u;
  uint32_t x0 = c0 + ks0;
  uint32_t x1 = c1 + ks1;
#define TF_R(r) { x0 += x1; x1 = rotl32(x1, r); x1 ^= x0; }
  TF_R(13) TF_R(15) TF_R(26) TF_R(6)
  x0 += ks1; x1 += ks2 + 1u;
  TF_R(17) TF_R(29) TF_R(16) TF_R(24)
  x0 += ks2; x1 += ks0 + 2u;
  TF_R(13) TF_R(15) TF_R(26) TF_R(6)
  x0 += ks0; x1 += ks1 + 3u;
  TF_R(17) TF_R(29) TF_R(16) TF_R(24)
  x0 += ks1; x1 += ks2 + 4u;
  TF_R(13) TF_R(15) TF_R(26) TF_R(6)
  x0 += ks2; x1 += ks0 + 5u;
#undef TF_R
  o0 = x0; o1 = x1;
}

// jax.random.uniform bit trick: [1,2) mantissa fill, minus 1
__device__ __forceinline__ float tf_uniform(uint32_t bits) {
  return __uint_as_float((bits >> 9) | 0x3f800000u) - 1.0f;
}

__device__ __forceinline__ float clip01(float v) {
  return fminf(fmaxf(v, 0.0f), 1.0f);
}

union HFrag {                      // whole-dword fragment assembly
  uint32_t u[8];
  v16h     h;
};

// Hash two consecutive flat indices; pack both time-halves' spikes as
// packed f16 pairs (one dword per fragment VGPR) -> no sub-register moves.
__device__ __forceinline__ void spike_pack2(uint32_t i0, float p0, float p1,
                                            uint32_t &d_lo, uint32_t &d_hi) {
  uint32_t a0, a1, b0, b1;
  threefry2x32(i0,      i0      + HALF_COUNT, a0, a1);
  threefry2x32(i0 + 1u, i0 + 1u + HALF_COUNT, b0, b1);
  d_lo = ((tf_uniform(a0) < p0) ? F16_ONE_LO : 0u) |
         ((tf_uniform(b0) < p1) ? F16_ONE_HI : 0u);
  d_hi = ((tf_uniform(a1) < p0) ? F16_ONE_LO : 0u) |
         ((tf_uniform(b1) < p1) ? F16_ONE_HI : 0u);
}

// ---------------------------------------------------------------------------
// Kernel 1: fused threefry spike-gen + WMMA GEMM.  Per wave:
//   M (16 rows) = 16 (t,b) pairs, t in [0,100)   -> c_lo tile
//                 same pairs at t+100 (hash lane1) -> c_hi tile
//   K           = feature n, chunks of 32, split into N_SEG segments
//   N (16 cols) = output unit h (0..2 live, rest zero-padded)
// A 16-bit 16x32 layout: lane g=0 -> K {0..7,16..23}, g=1 -> K {8..15,24..31}
// ---------------------------------------------------------------------------
__global__ __launch_bounds__(256)
void adasyn_cur_kernel(const float* __restrict__ x,
                       const float* __restrict__ W,
                       float* __restrict__ partial) {
  const int lane = threadIdx.x & 31;
  const int wid  = threadIdx.x >> 5;
  const int gw   = blockIdx.x * WAVES_PER_BLOCK + wid;
  const int w    = gw >> 3;            // pair-block  [0, 800)
  const int s    = gw & (N_SEG - 1);   // K segment   [0, 8)

  const int g  = lane >> 4;            // K-quarter group (A) / K-half (B)
  const int m  = lane & 15;            // A row owned by this lane
  const int nc = lane & 15;            // B/C column (= h)

  const int pair = w * 16 + m;
  const int bb   = pair & 127;
  const uint32_t base_i = (uint32_t)pair * (uint32_t)NFEAT;
  const float* xrow = x + bb * NFEAT;
  const bool   wcol = (nc < HOUT);
  const float* wrow = W + (wcol ? nc : 0) * NFEAT;

  v8f c_lo = {};   // cur[t,     b, h]
  v8f c_hi = {};   // cur[t+100, b, h]

  const int c_begin    = s * CHUNKS_PER_SEG;
  int c_end            = c_begin + CHUNKS_PER_SEG;
  if (c_end > CHUNKS) c_end = CHUNKS;
  const int c_full_end = (c_end < FULL_CHUNKS) ? c_end : FULL_CHUNKS;

  // ---- full chunks: no bounds checks, b128 vector loads -------------------
  for (int c = c_begin; c < c_full_end; ++c) {
    const int n0 = c * 32;
    __builtin_prefetch(xrow + n0 + 64, 0, 0);   // global_prefetch_b8

    // A: two contiguous 8-float runs of x for this lane's K-quarter
    const int k1 = n0 + 8 * g;
    const int k2 = k1 + 16;
    const float4 xa = *(const float4*)(xrow + k1);
    const float4 xb = *(const float4*)(xrow + k1 + 4);
    const float4 xc = *(const float4*)(xrow + k2);
    const float4 xd = *(const float4*)(xrow + k2 + 4);
    const float px[16] = { xa.x, xa.y, xa.z, xa.w, xb.x, xb.y, xb.z, xb.w,
                           xc.x, xc.y, xc.z, xc.w, xd.x, xd.y, xd.z, xd.w };
    HFrag alo, ahi;
#pragma unroll
    for (int v = 0; v < 8; ++v) {
      const int e0 = 2 * v;                       // e0,e0+1 sit in same run
      const int n  = (e0 < 8) ? (k1 + e0) : (k2 + e0 - 8);
      spike_pack2(base_i + (uint32_t)n,
                  clip01(px[e0]), clip01(px[e0 + 1]),
                  alo.u[v], ahi.u[v]);
    }

    // B: two contiguous 8-float runs of W for this lane's K-quarter (hi+lo f16)
    v16h b_h = {}, b_l = {};
    if (wcol) {
      const float* wp = wrow + n0 + 8 * g;
      const float4 w0 = *(const float4*)(wp);
      const float4 w1 = *(const float4*)(wp + 4);
      const float4 w2 = *(const float4*)(wp + 16);
      const float4 w3 = *(const float4*)(wp + 20);
      const float wv[16] = { w0.x, w0.y, w0.z, w0.w, w1.x, w1.y, w1.z, w1.w,
                             w2.x, w2.y, w2.z, w2.w, w3.x, w3.y, w3.z, w3.w };
#pragma unroll
      for (int e = 0; e < 16; ++e) {
        const _Float16 hi = (_Float16)wv[e];
        b_h[e] = hi;
        b_l[e] = (_Float16)(wv[e] - (float)hi);
      }
    }

    c_lo = __builtin_amdgcn_wmma_f32_16x16x32_f16(false, alo.h, false, b_h,
                                                  (short)0, c_lo, false, false);
    c_lo = __builtin_amdgcn_wmma_f32_16x16x32_f16(false, alo.h, false, b_l,
                                                  (short)0, c_lo, false, false);
    c_hi = __builtin_amdgcn_wmma_f32_16x16x32_f16(false, ahi.h, false, b_h,
                                                  (short)0, c_hi, false, false);
    c_hi = __builtin_amdgcn_wmma_f32_16x16x32_f16(false, ahi.h, false, b_l,
                                                  (short)0, c_hi, false, false);
  }

  // ---- tail chunk (c = 312, only segment 7): guarded scalar path ----------
  if (c_end > FULL_CHUNKS) {
    const int n0 = FULL_CHUNKS * 32;              // 9984
    v16h alo, ahi;
#pragma unroll
    for (int e = 0; e < 16; ++e) {
      const int K = (e < 8 ? e : e + 8) + 8 * g;
      const int n = n0 + K;
      _Float16 s0 = (_Float16)0.0f, s1 = (_Float16)0.0f;
      if (n < NFEAT) {
        const float p = clip01(xrow[n]);
        const uint32_t i = base_i + (uint32_t)n;
        uint32_t r0, r1;
        threefry2x32(i, i + HALF_COUNT, r0, r1);
        s0 = (tf_uniform(r0) < p) ? (_Float16)1.0f : (_Float16)0.0f;
        s1 = (tf_uniform(r1) < p) ? (_Float16)1.0f : (_Float16)0.0f;
      }
      alo[e] = s0;
      ahi[e] = s1;
    }
    v16h b_h = {}, b_l = {};
#pragma unroll
    for (int e = 0; e < 16; ++e) {
      const int n = n0 + (e < 8 ? e : e + 8) + 8 * g;
      if (wcol && n < NFEAT) {
        const float wv = wrow[n];
        const _Float16 hi = (_Float16)wv;
        b_h[e] = hi;
        b_l[e] = (_Float16)(wv - (float)hi);
      }
    }
    c_lo = __builtin_amdgcn_wmma_f32_16x16x32_f16(false, alo, false, b_h,
                                                  (short)0, c_lo, false, false);
    c_lo = __builtin_amdgcn_wmma_f32_16x16x32_f16(false, alo, false, b_l,
                                                  (short)0, c_lo, false, false);
    c_hi = __builtin_amdgcn_wmma_f32_16x16x32_f16(false, ahi, false, b_h,
                                                  (short)0, c_hi, false, false);
    c_hi = __builtin_amdgcn_wmma_f32_16x16x32_f16(false, ahi, false, b_l,
                                                  (short)0, c_hi, false, false);
  }

  // C/D 16x16 f32 layout: lane L, reg r -> row M = r + 8*(L>>4), col N = L%16
  if (wcol) {
#pragma unroll
    for (int r = 0; r < 8; ++r) {
      const int row = r + 8 * g;
      const int pr  = w * 16 + row;
      const int tt  = pr >> 7;
      const int bo  = pr & 127;
      partial[(((size_t)s * T_STEPS + tt) * BATCH + bo) * HOUT + nc] = c_lo[r];
      partial[(((size_t)s * T_STEPS + (tt + T_HALF)) * BATCH + bo) * HOUT + nc] = c_hi[r];
    }
  }
}

// ---------------------------------------------------------------------------
// Kernel 2: segment reduction + bias + 200-step adaptive-LIF scan.
// ---------------------------------------------------------------------------
__global__ __launch_bounds__(384)
void adasyn_scan_kernel(const float* __restrict__ partial,
                        const float* __restrict__ bias,
                        float* __restrict__ out) {
  const int idx = threadIdx.x;
  if (idx >= BATCH * HOUT) return;
  const int b = idx / HOUT;
  const int h = idx - b * HOUT;
  const float bh = bias[h];

  float syn = 0.0f, mem = 0.0f, thr = 1.0f;       // THR0
  float* spk_rec = out;
  float* mem_rec = out + (size_t)T_STEPS * BATCH * HOUT;

  for (int t = 0; t < T_STEPS; ++t) {
    float cur = bh;
#pragma unroll
    for (int s = 0; s < N_SEG; ++s)
      cur += partial[(((size_t)s * T_STEPS + t) * BATCH + b) * HOUT + h];

    syn = 0.9f * syn + cur;                       // ALPHA
    mem = 0.8f * mem + syn;                       // BETA
    const float spk = (mem > thr) ? 1.0f : 0.0f;
    mem -= spk * thr;                             // reset by subtraction
    thr = 0.9f * (thr - 1.0f) + 1.0f + spk;       // THR_DECAY/BASE/BETA_SCALE

    const size_t o = ((size_t)t * BATCH + b) * HOUT + h;
    spk_rec[o] = spk;
    mem_rec[o] = mem;
  }
}

extern "C" void kernel_launch(void* const* d_in, const int* in_sizes, int n_in,
                              void* d_out, int out_size, void* d_ws, size_t ws_size,
                              hipStream_t stream) {
  const float* x = (const float*)d_in[0];   // [128,100,100] -> [128,10000]
  const float* W = (const float*)d_in[1];   // [3,10000]
  const float* b = (const float*)d_in[2];   // [3]
  float* out     = (float*)d_out;           // spk_rec[200,128,3] ++ mem_rec[200,128,3]
  float* partial = (float*)d_ws;            // N_SEG*200*128*3 f32 = ~2.5 MB

  const int grid1 = (PAIRS / 16) * N_SEG / WAVES_PER_BLOCK;   // 800 blocks
  adasyn_cur_kernel<<<grid1, 256, 0, stream>>>(x, W, partial);
  adasyn_scan_kernel<<<1, 384, 0, stream>>>(partial, b, out);
}
